// EinconvNd_43989055045914
// MI455X (gfx1250) — matrix operand order repaired and checked by reading, not compiled, not run on hardware
//
#include <hip/hip_runtime.h>
#include <hip/hip_bf16.h>
#include <stdint.h>

// ---- problem constants (from reference) ----
constexpr int BATCH = 32;
constexpr int CIN   = 128;
constexpr int Hh    = 56;
constexpr int Ww    = 56;
constexpr int COUT  = 256;
constexpr int HW    = Hh * Ww;            // 3136
constexpr int CK    = CIN * 9;            // 1152 (im2col K)
constexpr int NTOT  = BATCH * HW;         // 100352
constexpr int W_ELEMS = COUT * CK;        // 294912
constexpr int X_ELEMS = BATCH * CIN * HW; // 12845056

// ---- tiling ----
constexpr int BM = 256;                   // full COUT tile (4 waves x 64)
constexpr int BN = 64;                    // spatial tile (2 waves x 32); 3136 % 64 == 0
constexpr int BK = 32;                    // one bf16 WMMA K
constexpr int LDA = BK + 8;               // padded bf16 row stride (80 B)
constexpr int LDB = BK + 8;
constexpr int NK  = CK / BK;              // 36 k-steps

typedef __attribute__((ext_vector_type(16))) __bf16 v16bf;
typedef __attribute__((ext_vector_type(8)))  __bf16 v8bf;
typedef __attribute__((ext_vector_type(4)))  __bf16 v4bf;
typedef __attribute__((ext_vector_type(8)))  float  v8f;
typedef __attribute__((ext_vector_type(4)))  float  v4f;
typedef __attribute__((ext_vector_type(4)))  unsigned int u32x4;
typedef __attribute__((ext_vector_type(8)))  int i32x8;
typedef __attribute__((ext_vector_type(4)))  int i32x4;

// =====================================================================
// TDM: DMA one 256x32 bf16 weight tile (row stride CK) into LDS with
// 16-DW interval / 4-DW padding -> 80-byte LDS rows (= LDA).
// D# layout per CDNA5 ISA ch.8 (group0 128b, group1 256b).
// =====================================================================
__device__ __forceinline__ void tdm_load_A(const __bf16* gsrc, __bf16* ldst) {
    const unsigned long long ga = (unsigned long long)(uintptr_t)gsrc;
    const unsigned lds = (unsigned)(uintptr_t)ldst;   // low 32b of flat = LDS offset

    u32x4 g0 = {
        1u,                                           // count=1, user descriptor
        lds,                                          // lds_addr [63:32]
        (unsigned)(ga & 0xffffffffu),                 // global_addr [95:64]
        (unsigned)((ga >> 32) & 0x01ffffffu) | (2u << 30) // addr [120:96] | type=2
    };
    i32x8 g1 = {
        (int)((1u << 16) |                            // data_size = 2 bytes
              (1u << 20) |                            // pad_enable
              (3u << 22) |                            // pad_interval: 16 DWORDs (64 B)
              (3u << 25)),                            // pad_amount: 4 DWORDs (16 B)
        (int)((unsigned)CK << 16),                    // tensor_dim0[15:0] @ bits 63:48
        (int)((unsigned)BM << 16),                    // dim0[31:16]=0, tensor_dim1[15:0]
        (int)((unsigned)BK << 16),                    // dim1[31:16]=0, tile_dim0=32
        (int)BM,                                      // tile_dim1=256, tile_dim2=0
        (int)CK,                                      // tensor_dim0_stride[31:0]
        0,                                            // stride[47:32]=0, dim1_stride lo=0
        0
    };
    i32x4 z = { 0, 0, 0, 0 };
#if defined(__clang_major__) && (__clang_major__ >= 23)
    i32x8 z8 = { 0, 0, 0, 0, 0, 0, 0, 0 };
    __builtin_amdgcn_tensor_load_to_lds(g0, g1, z, z, z8, 0);
#else
    __builtin_amdgcn_tensor_load_to_lds(g0, g1, z, z, 0);
#endif
}

// =====================================================================
// pre-pass: fp32 -> bf16 (vectorized, n multiple of 4)
// =====================================================================
__global__ __launch_bounds__(256)
void cvt_f32_bf16(const float* __restrict__ in, __bf16* __restrict__ op, int n4) {
    const int i = blockIdx.x * blockDim.x + threadIdx.x;
    if (i < n4) {
        v4f v = *(const v4f*)(in + 4 * i);
        v4bf o;
        #pragma unroll
        for (int j = 0; j < 4; ++j) o[j] = (__bf16)v[j];
        *(v4bf*)(op + 4 * i) = o;
    }
}

// =====================================================================
// main kernel, bf16 sources: TDM A-tiles + VALU im2col B-tiles
// =====================================================================
__global__ __launch_bounds__(256)
void einconv_wmma_tdm(const __bf16* __restrict__ xb16,
                      const __bf16* __restrict__ wb16,
                      const float* __restrict__ bias,
                      float* __restrict__ out) {
    __shared__ alignas(16) __bf16 As[2][BM * LDA];
    __shared__ alignas(16) __bf16 Bs[2][BN * LDB];

    const int tid  = threadIdx.x;
    const int lane = tid & 31;
    const int wave = tid >> 5;
    const int wm   = wave & 3;
    const int wn   = wave >> 2;

    const int tile_n = blockIdx.x * BN;
    const int bimg   = tile_n / HW;
    const int n_rem0 = tile_n % HW;

    const int b_kl = tid >> 3;             // 0..31 local k
    const int b_n0 = (tid & 7) * 8;        // 8 consecutive n

    int ohv[8], owv[8];
    #pragma unroll
    for (int u = 0; u < 8; ++u) {
        const int rem = n_rem0 + b_n0 + u;
        const int oh  = rem / Ww;
        ohv[u] = oh;
        owv[u] = rem - oh * Ww;
    }

    const __bf16* xb = xb16 + bimg * CIN * HW;
    __bf16 breg[8];

    auto gloadB = [&](int k0) {
        const int kg = k0 + b_kl;
        const int ci = kg / 9;
        const int rr = kg - ci * 9;
        const int kh = rr / 3;
        const int dh = kh - 1;
        const int dw = rr - kh * 3 - 1;
        const __bf16* xc = xb + ci * HW;
        #pragma unroll
        for (int u = 0; u < 8; ++u) {
            const int ih = ohv[u] + dh;
            const int iw = owv[u] + dw;
            const bool inb = (unsigned)ih < (unsigned)Hh && (unsigned)iw < (unsigned)Ww;
            breg[u] = inb ? xc[ih * Ww + iw] : (__bf16)0.0f;
        }
    };
    auto storeB = [&](int buf) {
        __bf16* bp = &Bs[buf][0];
        #pragma unroll
        for (int u = 0; u < 8; ++u)
            bp[(b_n0 + u) * LDB + b_kl] = breg[u];
    };

    const int am0 = wm * 64 + (lane & 15);
    const int ak  = (lane >> 4) * 8;
    const int bk  = (lane >> 4) * 8;
    const int bn0 = wn * 32 + (lane & 15);

    v8f acc[4][2];
    #pragma unroll
    for (int mi = 0; mi < 4; ++mi) { acc[mi][0] = (v8f)(0.0f); acc[mi][1] = (v8f)(0.0f); }

    auto mk16 = [](v8bf lo, v8bf hi) {
        return __builtin_shufflevector(lo, hi,
            0,1,2,3,4,5,6,7,8,9,10,11,12,13,14,15);
    };

    // ---- pipeline prologue ----
    if (wave == 0) tdm_load_A(wb16, &As[0][0]);
    gloadB(0);
    storeB(0);

    for (int ks = 0; ks < NK; ++ks) {
        if (ks + 1 < NK) gloadB((ks + 1) * BK);

        __builtin_amdgcn_s_wait_tensorcnt(0);   // wave0: A tile buf[ks&1] landed
        __syncthreads();                        // publish A (TDM) + B (ds) to all waves

        const __bf16* a_ = As[ks & 1];
        const __bf16* b_ = Bs[ks & 1];

        v16bf bf0 = mk16(*(const v8bf*)&b_[bn0 * LDB + bk],
                         *(const v8bf*)&b_[bn0 * LDB + bk + 16]);
        v16bf bf1 = mk16(*(const v8bf*)&b_[(bn0 + 16) * LDB + bk],
                         *(const v8bf*)&b_[(bn0 + 16) * LDB + bk + 16]);

        #pragma unroll
        for (int mi = 0; mi < 4; ++mi) {
            const int am = am0 + mi * 16;
            v16bf af = mk16(*(const v8bf*)&a_[am * LDA + ak],
                            *(const v8bf*)&a_[am * LDA + ak + 16]);
            acc[mi][0] = __builtin_amdgcn_wmma_f32_16x16x32_bf16(
                false, af, false, bf0, (short)0, acc[mi][0], false, false);
            acc[mi][1] = __builtin_amdgcn_wmma_f32_16x16x32_bf16(
                false, af, false, bf1, (short)0, acc[mi][1], false, false);
        }

        if (ks + 1 < NK) {
            storeB((ks + 1) & 1);
            if (wave == 0) tdm_load_A(wb16 + (ks + 1) * BK, &As[(ks + 1) & 1][0]);
        }
    }

    // ---- epilogue ----
    const int n_loc = lane & 15;
    #pragma unroll
    for (int mi = 0; mi < 4; ++mi) {
        const int m_base = wm * 64 + mi * 16 + ((lane >> 4) << 3);
        #pragma unroll
        for (int sub = 0; sub < 2; ++sub) {
            const int rem = n_rem0 + wn * 32 + sub * 16 + n_loc;
            float* op = out + (long)(bimg * COUT + m_base) * HW + rem;
            #pragma unroll
            for (int r = 0; r < 8; ++r)
                op[(long)r * HW] = acc[mi][sub][r] + bias[m_base + r];
        }
    }
}

// =====================================================================
// fallback (fp32 sources, no workspace): round-3 kernel
// =====================================================================
__global__ __launch_bounds__(256)
void einconv_wmma_f32src(const float* __restrict__ x,
                         const float* __restrict__ w,
                         const float* __restrict__ bias,
                         float* __restrict__ out) {
    __shared__ alignas(16) __bf16 As[2][BM * LDA];
    __shared__ alignas(16) __bf16 Bs[2][BN * LDB];

    const int tid  = threadIdx.x;
    const int lane = tid & 31;
    const int wave = tid >> 5;
    const int wm   = wave & 3;
    const int wn   = wave >> 2;

    const int tile_n = blockIdx.x * BN;
    const int bimg   = tile_n / HW;
    const int n_rem0 = tile_n % HW;

    const int a_row = tid;
    const int b_kl  = tid >> 3;
    const int b_n0  = (tid & 7) * 8;

    int ohv[8], owv[8];
    #pragma unroll
    for (int u = 0; u < 8; ++u) {
        const int rem = n_rem0 + b_n0 + u;
        const int oh  = rem / Ww;
        ohv[u] = oh;
        owv[u] = rem - oh * Ww;
    }

    const float* xb   = x + bimg * CIN * HW;
    const float* wrow = w + a_row * CK;

    v4f   areg[8];
    float breg[8];

    auto gload = [&](int k0) {
        const float* wp = wrow + k0;
        __builtin_prefetch(wp + BK, 0, 0);
        #pragma unroll
        for (int i = 0; i < 8; ++i) areg[i] = *(const v4f*)(wp + 4 * i);
        const int kg = k0 + b_kl;
        const int ci = kg / 9;
        const int rr = kg - ci * 9;
        const int kh = rr / 3;
        const int dh = kh - 1;
        const int dw = rr - kh * 3 - 1;
        const float* xc = xb + ci * HW;
        #pragma unroll
        for (int u = 0; u < 8; ++u) {
            const int ih = ohv[u] + dh;
            const int iw = owv[u] + dw;
            const bool inb = (unsigned)ih < (unsigned)Hh && (unsigned)iw < (unsigned)Ww;
            breg[u] = inb ? xc[ih * Ww + iw] : 0.0f;
        }
    };
    auto lstore = [&](int buf) {
        __bf16* ap = &As[buf][a_row * LDA];
        #pragma unroll
        for (int q = 0; q < 4; ++q) {
            v8bf p;
            #pragma unroll
            for (int i = 0; i < 4; ++i) {
                p[i]     = (__bf16)areg[2 * q][i];
                p[i + 4] = (__bf16)areg[2 * q + 1][i];
            }
            *(v8bf*)(ap + 8 * q) = p;
        }
        __bf16* bp = &Bs[buf][0];
        #pragma unroll
        for (int u = 0; u < 8; ++u)
            bp[(b_n0 + u) * LDB + b_kl] = (__bf16)breg[u];
    };

    const int am0 = wm * 64 + (lane & 15);
    const int ak  = (lane >> 4) * 8;
    const int bk  = (lane >> 4) * 8;
    const int bn0 = wn * 32 + (lane & 15);

    v8f acc[4][2];
    #pragma unroll
    for (int mi = 0; mi < 4; ++mi) { acc[mi][0] = (v8f)(0.0f); acc[mi][1] = (v8f)(0.0f); }

    auto mk16 = [](v8bf lo, v8bf hi) {
        return __builtin_shufflevector(lo, hi,
            0,1,2,3,4,5,6,7,8,9,10,11,12,13,14,15);
    };

    gload(0);
    lstore(0);
    for (int ks = 0; ks < NK; ++ks) {
        if (ks + 1 < NK) gload((ks + 1) * BK);
        __syncthreads();
        const __bf16* a_ = As[ks & 1];
        const __bf16* b_ = Bs[ks & 1];
        v16bf bf0 = mk16(*(const v8bf*)&b_[bn0 * LDB + bk],
                         *(const v8bf*)&b_[bn0 * LDB + bk + 16]);
        v16bf bf1 = mk16(*(const v8bf*)&b_[(bn0 + 16) * LDB + bk],
                         *(const v8bf*)&b_[(bn0 + 16) * LDB + bk + 16]);
        #pragma unroll
        for (int mi = 0; mi < 4; ++mi) {
            const int am = am0 + mi * 16;
            v16bf af = mk16(*(const v8bf*)&a_[am * LDA + ak],
                            *(const v8bf*)&a_[am * LDA + ak + 16]);
            acc[mi][0] = __builtin_amdgcn_wmma_f32_16x16x32_bf16(
                false, af, false, bf0, (short)0, acc[mi][0], false, false);
            acc[mi][1] = __builtin_amdgcn_wmma_f32_16x16x32_bf16(
                false, af, false, bf1, (short)0, acc[mi][1], false, false);
        }
        if (ks + 1 < NK) lstore((ks + 1) & 1);
    }

    const int n_loc = lane & 15;
    #pragma unroll
    for (int mi = 0; mi < 4; ++mi) {
        const int m_base = wm * 64 + mi * 16 + ((lane >> 4) << 3);
        #pragma unroll
        for (int sub = 0; sub < 2; ++sub) {
            const int rem = n_rem0 + wn * 32 + sub * 16 + n_loc;
            float* op = out + (long)(bimg * COUT + m_base) * HW + rem;
            #pragma unroll
            for (int r = 0; r < 8; ++r)
                op[(long)r * HW] = acc[mi][sub][r] + bias[m_base + r];
        }
    }
}

extern "C" void kernel_launch(void* const* d_in, const int* in_sizes, int n_in,
                              void* d_out, int out_size, void* d_ws, size_t ws_size,
                              hipStream_t stream) {
    (void)in_sizes; (void)n_in; (void)out_size;
    const float* x    = (const float*)d_in[0];
    const float* w    = (const float*)d_in[1];
    const float* bias = (const float*)d_in[2];
    float* out        = (float*)d_out;

    const size_t ws_needed = (size_t)(W_ELEMS + X_ELEMS) * sizeof(__bf16);
    dim3 grid(NTOT / BN, 1);

    if (d_ws != nullptr && ws_size >= ws_needed) {
        __bf16* wb16 = (__bf16*)d_ws;                  // [COUT][CK]
        __bf16* xb16 = wb16 + W_ELEMS;                 // [B][CIN][H][W]
        cvt_f32_bf16<<<(W_ELEMS / 4 + 255) / 256, 256, 0, stream>>>(w, wb16, W_ELEMS / 4);
        cvt_f32_bf16<<<(X_ELEMS / 4 + 255) / 256, 256, 0, stream>>>(x, xb16, X_ELEMS / 4);
        einconv_wmma_tdm<<<grid, 256, 0, stream>>>(xb16, wb16, bias, out);
    } else {
        einconv_wmma_f32src<<<grid, 256, 0, stream>>>(x, w, bias, out);
    }
}
